// DeformConvPack_d_48052094108375
// MI455X (gfx1250) — matrix operand order repaired and checked by reading, not compiled, not run on hardware
//
#include <hip/hip_runtime.h>

#define B_    2
#define CIN_  32
#define COUT_ 64
#define T_    8
#define H_    64
#define W_    64
#define KPTS_ 27
#define THW_  (T_ * H_ * W_)          // 32768
#define KTOT_ (CIN_ * KPTS_)          // 864  (K ordering: k = kpt*32 + cin)
#define OFFCH_ 54
#define NTILE_ 16                     // output positions per block (along W)
#define NSTEPS_ 27                    // KTOT_ / 32 WMMA k-steps
#define LDS_STRIDE_ 872               // padded bf16 row stride: 16B-aligned, bank-conflict-free
#define FRAG_ELEMS_ (4 * NSTEPS_ * 32 * 16)          // 55296 bf16 per pre-swizzled weight buffer
#define OFFS_BYTES_ ((size_t)B_ * OFFCH_ * THW_ * sizeof(float))

typedef __attribute__((ext_vector_type(16))) __bf16 v16bf;
typedef __attribute__((ext_vector_type(8)))  float  v8f;

union Frag16 { uint4 q[2]; v16bf v; };
union Pack8  { __bf16 h[8]; uint4 q; };

// ---------------------------------------------------------------------------
// Pre-swizzle weights into per-lane WMMA A-fragments (bf16).
// A-matrix 16x32 lane layout (ISA 7.12.2): lane = half*16 + row,
// element e -> klocal = half*16 + e (contiguous k order); k = kstep*32 + klocal.
// K ordering chosen as k = kpt*32 + cin so kpt == kstep, cin == klocal.
// ---------------------------------------------------------------------------
__global__ __launch_bounds__(256) void prep_frags(
    const float* __restrict__ weight, const float* __restrict__ offw,
    __bf16* __restrict__ wfrag, __bf16* __restrict__ owfrag)
{
    int tid = blockIdx.x * 256 + threadIdx.x;
    if (tid >= 2 * FRAG_ELEMS_) return;
    int e    = tid & 15;
    int lane = (tid >> 4) & 31;
    int ks   = (tid >> 9) % NSTEPS_;
    int ct   = (tid / (512 * NSTEPS_)) & 3;
    int buf  = tid / (512 * NSTEPS_ * 4);
    int row  = lane & 15;
    int half = lane >> 4;
    int klocal = half * 16 + e;
    int cin = klocal;
    int kpt = ks;
    int cout = ct * 16 + row;
    float v;
    if (buf == 0) {
        v = weight[((size_t)cout * CIN_ + cin) * KPTS_ + kpt];
    } else {
        v = (cout < OFFCH_) ? offw[((size_t)cout * CIN_ + cin) * KPTS_ + kpt] : 0.0f;
    }
    __bf16* dst = (buf == 0) ? wfrag : owfrag;
    dst[(((size_t)ct * NSTEPS_ + ks) * 32 + lane) * 16 + e] = (__bf16)v;
}

// ---------------------------------------------------------------------------
// Offset conv (54ch 3x3x3, zero pad) as im2col + bf16 WMMA GEMM.
// Block: 16 W-consecutive positions x 64(54) channels, 4 waves.
// ---------------------------------------------------------------------------
__global__ __launch_bounds__(128) void offset_conv_wmma(
    const float* __restrict__ x, const __bf16* __restrict__ owfrag,
    const float* __restrict__ offset_bias, float* __restrict__ offs)
{
    __shared__ __bf16 sP[NTILE_ * LDS_STRIDE_];
    int bid = blockIdx.x;
    int n16 = bid & 3;
    int h   = (bid >> 2) & 63;
    int t   = (bid >> 8) & 7;
    int b   = bid >> 11;
    int w0  = n16 * 16;

    // Phase 1: build patch matrix [k=864][n=16] in LDS; 8-channel packed
    // bf16 stores -> one ds_store_b128 per 8 samples.
    for (int task = threadIdx.x; task < KPTS_ * NTILE_; task += 128) {
        int n = task & 15, kpt = task >> 4;
        int kt = kpt / 9, kr = kpt % 9, kh = kr / 3, kw = kr % 3;
        int tt = t + kt - 1, hh = h + kh - 1, ww = w0 + n + kw - 1;
        bool valid = ((unsigned)tt < T_) && ((unsigned)hh < H_) && ((unsigned)ww < W_);
        const float* xb = x + (size_t)b * CIN_ * THW_ +
                          (valid ? (tt * H_ * W_ + hh * W_ + ww) : 0);
        __bf16* dst = &sP[n * LDS_STRIDE_ + kpt * 32];
        #pragma unroll
        for (int c0 = 0; c0 < CIN_; c0 += 8) {
            Pack8 pk;
            #pragma unroll
            for (int j = 0; j < 8; ++j) {
                float v = valid ? xb[(size_t)(c0 + j) * THW_] : 0.0f;
                pk.h[j] = (__bf16)v;
            }
            *(uint4*)&dst[c0] = pk.q;   // 16B-aligned: kpt*64 + n*1744
        }
    }
    __syncthreads();

    // Phase 2: one 16x16 f32 tile per wave, 27 WMMA k-steps.
    int lane = threadIdx.x & 31;
    int ct   = threadIdx.x >> 5;
    int n    = lane & 15;
    int half = lane >> 4;

    v8f acc;
    #pragma unroll
    for (int r = 0; r < 8; ++r) {
        int ch = ct * 16 + r + 8 * half;
        acc[r] = (ch < OFFCH_) ? offset_bias[ch] : 0.0f;
    }

    const char*  abase = (const char*)owfrag + ((size_t)ct * NSTEPS_ * 32 + lane) * 32;
    const __bf16* brow = &sP[n * LDS_STRIDE_ + half * 16];
    for (int ks = 0; ks < NSTEPS_; ++ks) {
        Frag16 fa, fb;
        const char* ap = abase + (size_t)ks * 32 * 32;
        fa.q[0] = *(const uint4*)ap;
        fa.q[1] = *(const uint4*)(ap + 16);
        const __bf16* bp = brow + ks * 32;
        fb.q[0] = *(const uint4*)bp;
        fb.q[1] = *(const uint4*)(bp + 8);
        acc = __builtin_amdgcn_wmma_f32_16x16x32_bf16(false, fa.v, false, fb.v,
                                                      (short)0, acc, false, false);
    }

    // Phase 3: write the 54 real channels (coalesced along W).
    int spatial = t * H_ * W_ + h * W_ + w0 + n;
    #pragma unroll
    for (int r = 0; r < 8; ++r) {
        int ch = ct * 16 + r + 8 * half;
        if (ch < OFFCH_)
            offs[((size_t)b * OFFCH_ + ch) * THW_ + spatial] = acc[r];
    }
}

// ---------------------------------------------------------------------------
// Main deformable conv: bilinear sampling (d_t == 0 in the reference, so the
// trilinear T-axis collapses: 4 taps instead of 8) into LDS, then bf16 WMMA.
// ---------------------------------------------------------------------------
__global__ __launch_bounds__(128) void deform_conv_wmma(
    const float* __restrict__ x, const float* __restrict__ offs,
    const __bf16* __restrict__ wfrag, const float* __restrict__ bias,
    float* __restrict__ out)
{
    __shared__ __bf16 sP[NTILE_ * LDS_STRIDE_];
    int bid = blockIdx.x;
    int n16 = bid & 3;
    int h   = (bid >> 2) & 63;
    int t   = (bid >> 8) & 7;
    int b   = bid >> 11;
    int w0  = n16 * 16;

    // Phase 1: sampled patch XS[k=864][n=16] in LDS.
    for (int task = threadIdx.x; task < KPTS_ * NTILE_; task += 128) {
        int n = task & 15, kpt = task >> 4;
        int kt = kpt / 9, kr = kpt % 9, kh = kr / 3, kw = kr % 3;
        int it = t + kt - 1;
        bool vt = (unsigned)it < T_;
        int sp = t * H_ * W_ + h * W_ + (w0 + n);
        float dh = offs[((size_t)b * OFFCH_ + kpt * 2    ) * THW_ + sp];
        float dw = offs[((size_t)b * OFFCH_ + kpt * 2 + 1) * THW_ + sp];
        float ph = (float)(h + kh - 1) + dh;
        float pw = (float)(w0 + n + kw - 1) + dw;
        float fh = floorf(ph), fw = floorf(pw);
        float rh = ph - fh,    rw = pw - fw;
        int ih0 = (int)fh, iw0 = (int)fw;
        int ih1 = ih0 + 1, iw1 = iw0 + 1;
        float mt  = vt ? 1.0f : 0.0f;
        float mh0 = ((unsigned)ih0 < H_) ? mt : 0.0f;
        float mh1 = ((unsigned)ih1 < H_) ? mt : 0.0f;
        float mw0 = ((unsigned)iw0 < W_) ? 1.0f : 0.0f;
        float mw1 = ((unsigned)iw1 < W_) ? 1.0f : 0.0f;
        float s00 = (1.0f - rh) * (1.0f - rw) * mh0 * mw0;
        float s01 = (1.0f - rh) * rw          * mh0 * mw1;
        float s10 = rh * (1.0f - rw)          * mh1 * mw0;
        float s11 = rh * rw                   * mh1 * mw1;
        int itc = min(max(it, 0), T_ - 1);
        int ch0 = min(max(ih0, 0), H_ - 1), ch1 = min(max(ih1, 0), H_ - 1);
        int cw0 = min(max(iw0, 0), W_ - 1), cw1 = min(max(iw1, 0), W_ - 1);
        int i00 = (itc * H_ + ch0) * W_ + cw0;
        int i01 = (itc * H_ + ch0) * W_ + cw1;
        int i10 = (itc * H_ + ch1) * W_ + cw0;
        int i11 = (itc * H_ + ch1) * W_ + cw1;
        const float* xb = x + (size_t)b * CIN_ * THW_;
        __bf16* dst = &sP[n * LDS_STRIDE_ + kpt * 32];
        #pragma unroll
        for (int c0 = 0; c0 < CIN_; c0 += 8) {
            Pack8 pk;
            #pragma unroll
            for (int j = 0; j < 8; ++j) {
                const float* p = xb + (size_t)(c0 + j) * THW_;
                float v = s00 * p[i00] + s01 * p[i01] + s10 * p[i10] + s11 * p[i11];
                pk.h[j] = (__bf16)v;
            }
            *(uint4*)&dst[c0] = pk.q;   // one ds_store_b128 per 8 channels
        }
    }
    __syncthreads();

    // Phase 2: 16x16 f32 accumulator per wave, 27 bf16 WMMA k-steps.
    int lane = threadIdx.x & 31;
    int ct   = threadIdx.x >> 5;
    int n    = lane & 15;
    int half = lane >> 4;

    v8f acc;
    #pragma unroll
    for (int r = 0; r < 8; ++r)
        acc[r] = bias[ct * 16 + r + 8 * half];

    const char*  abase = (const char*)wfrag + ((size_t)ct * NSTEPS_ * 32 + lane) * 32;
    const __bf16* brow = &sP[n * LDS_STRIDE_ + half * 16];
    for (int ks = 0; ks < NSTEPS_; ++ks) {
        Frag16 fa, fb;
        const char* ap = abase + (size_t)ks * 32 * 32;
        fa.q[0] = *(const uint4*)ap;
        fa.q[1] = *(const uint4*)(ap + 16);
        const __bf16* bp = brow + ks * 32;
        fb.q[0] = *(const uint4*)bp;
        fb.q[1] = *(const uint4*)(bp + 8);
        acc = __builtin_amdgcn_wmma_f32_16x16x32_bf16(false, fa.v, false, fb.v,
                                                      (short)0, acc, false, false);
    }

    // Phase 3: write out[b][cout][t][h][w], coalesced along W across lanes.
    int spatial = t * H_ * W_ + h * W_ + w0 + n;
    float* ob = out + ((size_t)b * COUT_ + ct * 16) * THW_ + spatial;
    #pragma unroll
    for (int r = 0; r < 8; ++r)
        ob[(size_t)(r + 8 * half) * THW_] = acc[r];
}

// ---------------------------------------------------------------------------
extern "C" void kernel_launch(void* const* d_in, const int* in_sizes, int n_in,
                              void* d_out, int out_size, void* d_ws, size_t ws_size,
                              hipStream_t stream)
{
    const float* x      = (const float*)d_in[0];
    const float* weight = (const float*)d_in[1];
    const float* bias   = (const float*)d_in[2];
    const float* offw   = (const float*)d_in[3];
    const float* offb   = (const float*)d_in[4];
    float* out  = (float*)d_out;

    float*  offs   = (float*)d_ws;                              // 14,155,776 B
    __bf16* wfrag  = (__bf16*)((char*)d_ws + OFFS_BYTES_);      //    110,592 B
    __bf16* owfrag = wfrag + FRAG_ELEMS_;                       //    110,592 B

    prep_frags<<<(2 * FRAG_ELEMS_ + 255) / 256, 256, 0, stream>>>(
        weight, offw, wfrag, owfrag);

    int nblk = B_ * T_ * H_ * (W_ / NTILE_);   // 4096
    offset_conv_wmma<<<nblk, 128, 0, stream>>>(x, owfrag, offb, offs);
    deform_conv_wmma<<<nblk, 128, 0, stream>>>(x, offs, wfrag, bias, out);
}
